// FactoredLSTM_88184268521993
// MI455X (gfx1250) — matrix operand order repaired and checked by reading, not compiled, not run on hardware
//
#include <hip/hip_runtime.h>
#include <hip/hip_bf16.h>

typedef __attribute__((ext_vector_type(16))) __bf16 v16bf;
typedef __attribute__((ext_vector_type(8)))  __bf16 v8bf;
typedef __attribute__((ext_vector_type(8)))  float  v8f;

#define NB   64      // batch
#define NT   32      // time steps
#define NE   512     // embed
#define NF   512     // factored dim
#define NH   1024    // hidden
#define NV   10000   // vocab

// ---------------------------------------------------------------------------
// f32 -> bf16 conversion (grid-stride)
// ---------------------------------------------------------------------------
__global__ void cvt_bf16(const float* __restrict__ s, __bf16* __restrict__ d, size_t n) {
    size_t i = (size_t)blockIdx.x * blockDim.x + threadIdx.x;
    size_t stride = (size_t)gridDim.x * blockDim.x;
    for (; i < n; i += stride) d[i] = (__bf16)s[i];
}

// ---------------------------------------------------------------------------
// Build x (B*T x E) in bf16: row b*32+t = image_features[b] (t==0) else
// B_emb[captions[b, t-1]]
// ---------------------------------------------------------------------------
__global__ void build_x(const int* __restrict__ captions,
                        const float* __restrict__ imgf,
                        const float* __restrict__ Bemb,
                        __bf16* __restrict__ xbf) {
    size_t i = (size_t)blockIdx.x * blockDim.x + threadIdx.x;   // < 2048*512
    int   e   = (int)(i & (NE - 1));
    int   row = (int)(i >> 9);
    int   t   = row & (NT - 1);
    int   b   = row >> 5;
    float v;
    if (t == 0) v = imgf[(size_t)b * NE + e];
    else        v = Bemb[(size_t)captions[b * NT + (t - 1)] * NE + e];
    xbf[i] = (__bf16)v;
}

// ---------------------------------------------------------------------------
// Init recurrent state: h -> bf16, c -> f32 workspace copy
// ---------------------------------------------------------------------------
__global__ void init_state(const float* __restrict__ h0, const float* __restrict__ c0,
                           __bf16* __restrict__ hbf, float* __restrict__ c) {
    int i = blockIdx.x * blockDim.x + threadIdx.x;              // < 64*1024
    hbf[i] = (__bf16)h0[i];
    c[i]   = c0[i];
}

// ---------------------------------------------------------------------------
// Register-blocked WMMA GEMM:
//   D[m,n] = sum_k A[m,k] * W[n,k] + bias[n] (+ Add[m,n])
//   A : (M x K) bf16, row stride lda
//   W : (N x K) bf16, row stride K  (out-dim-major => B fragment w/o transpose)
//   Each wave computes BM x BN 16x16 tiles (A frags reused across BN, B frags
//   across BM -> (2*BM + 2*BN) b128 loads per (BM*BN) WMMAs per K-step).
//   4 waves per block, stacked along N. K multiple of 32, M multiple of 16*BM.
//   Fragment layouts as loaded here (contiguous k, same for A and B):
//     A lane L: M=L%16, VGPR v:   K=16*(L/16)+2v..
//     B lane L: N=L%16, VGPR v:   K=16*(L/16)+2v..   (16 contiguous K per half)
//     C/D VGPR r: M = r + 8*(L/16), N = L%16
// ---------------------------------------------------------------------------
template <int BM, int BN, bool OUT_BF16, bool HAS_ADD>
__global__ __launch_bounds__(128)
void wmma_gemm(const __bf16* __restrict__ A, int lda,
               const __bf16* __restrict__ W,
               const float* __restrict__ bias,
               const float* __restrict__ Add, long ldadd,
               void* __restrict__ D, long ldd,
               int N, int K) {
    const int lane = threadIdx.x & 31;
    const int wave = threadIdx.x >> 5;
    const int l16  = lane & 15;
    const int half = lane >> 4;
    const int tn0  = (blockIdx.x * 4 + wave) * BN;   // first N tile of this wave
    const int tm0  = blockIdx.y * BM;                // first M tile
    if (tn0 * 16 >= N) return;                       // wave-uniform guard

    const __bf16* pa[BM];
#pragma unroll
    for (int i = 0; i < BM; ++i)
        pa[i] = A + (size_t)((tm0 + i) * 16 + l16) * lda;

    bool nvalid[BN];
    const __bf16* pw[BN];
#pragma unroll
    for (int j = 0; j < BN; ++j) {
        int tnj = tn0 + j;
        nvalid[j] = (tnj * 16 < N);
        if (!nvalid[j]) tnj = tn0;                   // clamp: loads stay in-bounds
        pw[j] = W + (size_t)(tnj * 16 + l16) * K;
    }

    v8f acc[BM][BN] = {};
    for (int k0 = 0; k0 < K; k0 += 32) {
        v16bf af[BM], bf[BN];
#pragma unroll
        for (int i = 0; i < BM; ++i) {
            af[i] = *(const v16bf*)(pa[i] + k0 + 16 * half);        // K = 16h..+15
        }
#pragma unroll
        for (int j = 0; j < BN; ++j)
            bf[j] = *(const v16bf*)(pw[j] + k0 + 16 * half);        // K = 16h..+15
#pragma unroll
        for (int i = 0; i < BM; ++i)
#pragma unroll
            for (int j = 0; j < BN; ++j)
                acc[i][j] = __builtin_amdgcn_wmma_f32_16x16x32_bf16(
                    false, af[i], false, bf[j], (short)0, acc[i][j], false, false);
    }

#pragma unroll
    for (int j = 0; j < BN; ++j) {
        if (!nvalid[j]) continue;                    // wave-uniform
        const int   n  = (tn0 + j) * 16 + l16;
        const float bv = bias[n];
#pragma unroll
        for (int i = 0; i < BM; ++i) {
#pragma unroll
            for (int r = 0; r < 8; ++r) {
                const int m = (tm0 + i) * 16 + 8 * half + r;
                float v = acc[i][j][r] + bv;
                if constexpr (HAS_ADD) v += Add[(size_t)m * ldadd + n];
                if constexpr (OUT_BF16) ((__bf16*)D)[(size_t)m * ldd + n] = (__bf16)v;
                else                    ((float*)D)[(size_t)m * ldd + n] = v;
            }
        }
    }
}

// ---------------------------------------------------------------------------
// LSTM cell elementwise: z (64 x 4096) = [i|f|o|ct] preactivations
// ---------------------------------------------------------------------------
__global__ void lstm_elem(const float* __restrict__ z, float* __restrict__ c,
                          __bf16* __restrict__ hbf) {
    int i = blockIdx.x * blockDim.x + threadIdx.x;              // < 64*1024
    int h = i & (NH - 1);
    int b = i >> 10;
    const float* zb = z + (size_t)b * (4 * NH);
    float zi = zb[h], zf = zb[NH + h], zo = zb[2 * NH + h], zc = zb[3 * NH + h];
    float it = 1.f / (1.f + __expf(-zi));
    float ft = 1.f / (1.f + __expf(-zf));
    float ot = 1.f / (1.f + __expf(-zo));
    float ct = tanhf(zc);
    float cn = ft * c[i] + it * ct;
    float hn = ot * cn;                 // reference: h = o * c_new (no tanh)
    c[i]   = cn;
    hbf[i] = (__bf16)hn;
}

// ---------------------------------------------------------------------------
extern "C" void kernel_launch(void* const* d_in, const int* in_sizes, int n_in,
                              void* d_out, int out_size, void* d_ws, size_t ws_size,
                              hipStream_t stream) {
    const int*   captions = (const int*)d_in[0];
    const float* imgf     = (const float*)d_in[1];
    const float* Bemb     = (const float*)d_in[2];
    const float* V_W      = (const float*)d_in[3];
    const float* V_b      = (const float*)d_in[4];
    const float* S_W      = (const float*)d_in[5];
    const float* S_b      = (const float*)d_in[6];
    const float* U_W      = (const float*)d_in[7];
    const float* U_b      = (const float*)d_in[8];
    const float* W_W      = (const float*)d_in[9];
    const float* W_b      = (const float*)d_in[10];
    const float* C_W      = (const float*)d_in[11];
    const float* C_b      = (const float*)d_in[12];
    const float* h0       = (const float*)d_in[13];
    const float* c0       = (const float*)d_in[14];
    float*       out      = (float*)d_out;

    // ---- workspace carve-up (all chunks multiple of 256B) ----
    char* w = (char*)d_ws;
    auto carve = [&](size_t bytes) { char* p = w; w += (bytes + 255) & ~(size_t)255; return p; };
    const size_t M  = (size_t)NB * NT;        // 2048
    __bf16* xbf  = (__bf16*)carve(M * NE * 2);
    __bf16* VWb  = (__bf16*)carve((size_t)4 * NF * NE * 2);
    __bf16* SWb  = (__bf16*)carve((size_t)4 * NF * NF * 2);
    __bf16* UWb  = (__bf16*)carve((size_t)4 * NH * NF * 2);
    __bf16* WWb  = (__bf16*)carve((size_t)4 * NH * NH * 2);
    __bf16* CWb  = (__bf16*)carve((size_t)NV * NH * 2);
    __bf16* g1   = (__bf16*)carve(M * 4 * NF * 2);
    __bf16* g2   = (__bf16*)carve(M * 4 * NF * 2);
    float*  g3   = (float*) carve(M * 4 * NH * 4);
    float*  zbuf = (float*) carve((size_t)NB * 4 * NH * 4);
    float*  cst  = (float*) carve((size_t)NB * NH * 4);
    __bf16* hbf  = (__bf16*)carve((size_t)NB * NH * 2);

    auto cvt = [&](const float* s, __bf16* d, size_t n) {
        int grid = (int)((n + 255) / 256); if (grid > 2048) grid = 2048;
        cvt_bf16<<<grid, 256, 0, stream>>>(s, d, n);
    };
    cvt(V_W, VWb, (size_t)4 * NF * NE);
    cvt(S_W, SWb, (size_t)4 * NF * NF);
    cvt(U_W, UWb, (size_t)4 * NH * NF);
    cvt(W_W, WWb, (size_t)4 * NH * NH);
    cvt(C_W, CWb, (size_t)NV * NH);

    build_x<<<(int)(M * NE / 256), 256, 0, stream>>>(captions, imgf, Bemb, xbf);
    init_state<<<NB * NH / 256, 256, 0, stream>>>(h0, c0, hbf, cst);

    auto cdiv = [](int a, int b) { return (a + b - 1) / b; };
    // grid.x = ceil(#Ntiles / BN / 4waves), grid.y = Mtiles / BM

    // ---- g1 = x @ V_W^T + V_b : M=2048, N=2048 (4 gates folded), K=512 ----
    wmma_gemm<2, 2, true, false>
        <<<dim3(cdiv(cdiv(2048 / 16, 2), 4), (int)(M / 32)), 128, 0, stream>>>(
        xbf, NE, VWb, V_b, nullptr, 0, g1, 4 * NF, 4 * NF, NE);

    // ---- g2[g] = g1[g] @ S_W[g]^T + S_b[g] : block-diagonal over gates ----
    for (int g = 0; g < 4; ++g)
        wmma_gemm<2, 2, true, false>
            <<<dim3(cdiv(cdiv(NF / 16, 2), 4), (int)(M / 32)), 128, 0, stream>>>(
            g1 + g * NF, 4 * NF, SWb + (size_t)g * NF * NF, S_b + g * NF,
            nullptr, 0, g2 + g * NF, 4 * NF, NF, NF);

    // ---- g3[g] = g2[g] @ U_W[g]^T + U_b[g] : f32 out (M x 4096) ----
    for (int g = 0; g < 4; ++g)
        wmma_gemm<2, 2, false, false>
            <<<dim3(cdiv(cdiv(NH / 16, 2), 4), (int)(M / 32)), 128, 0, stream>>>(
            g2 + g * NF, 4 * NF, UWb + (size_t)g * NH * NF, U_b + g * NH,
            nullptr, 0, g3 + g * NH, 4 * NH, NH, NF);

    // ---- sequential scan: 32 steps, 3 kernels each (stream-ordered) ----
    for (int t = 0; t < NT; ++t) {
        // z = h @ W_W^T + W_b + g3[:,t,:]   (M=64, N=4096, K=1024)
        // g3 row for (b,t) is b*32+t -> Add base g3 + t*4096, row stride 32*4096
        wmma_gemm<1, 2, false, true>
            <<<dim3(cdiv(cdiv(4 * NH / 16, 2), 4), NB / 16), 128, 0, stream>>>(
            hbf, NH, WWb, W_b, g3 + (size_t)t * 4 * NH, (long)NT * 4 * NH,
            zbuf, 4 * NH, 4 * NH, NH);

        lstm_elem<<<NB * NH / 256, 256, 0, stream>>>(zbuf, cst, hbf);

        // logits = h_new @ C_W^T + C_b -> out[b, t, :]  (M=64, N=10000, K=1024)
        wmma_gemm<1, 2, false, false>
            <<<dim3(cdiv(cdiv(cdiv(NV, 16), 2), 4), NB / 16), 128, 0, stream>>>(
            hbf, NH, CWb, C_b, nullptr, 0,
            out + (size_t)t * NV, (long)NT * NV, NV, NH);
    }
}
